// gat_5136780886428
// MI455X (gfx1250) — matrix operand order repaired and checked by reading, not compiled, not run on hardware
//
#include <hip/hip_runtime.h>
#include <hip/hip_bf16.h>
#include <cstddef>

// Problem constants (from reference)
#define BB 4
#define NN 4096
#define FF 256
#define UU 128
#define NEGV (-1.0e9f)

typedef __attribute__((ext_vector_type(16))) __bf16 v16bf;
typedef __attribute__((ext_vector_type(8)))  __bf16 v8bf;
typedef __attribute__((ext_vector_type(8)))  float  v8f;

static __device__ __forceinline__ unsigned short f32_to_bf16(float f) {
    unsigned int u = __float_as_uint(f);
    u += 0x7FFFu + ((u >> 16) & 1u);   // round-to-nearest-even
    return (unsigned short)(u >> 16);
}

static __device__ __forceinline__ v16bf cat16(v8bf lo, v8bf hi) {
    return __builtin_shufflevector(lo, hi, 0,1,2,3,4,5,6,7,8,9,10,11,12,13,14,15);
}

// ---------------------------------------------------------------- f32 -> bf16
__global__ void cvt_bf16_kernel(const float* __restrict__ src,
                                unsigned short* __restrict__ dst, int n) {
    for (int i = blockIdx.x * blockDim.x + threadIdx.x; i < n;
         i += gridDim.x * blockDim.x) {
        dst[i] = f32_to_bf16(src[i]);
    }
}

// ------------------------------------------------- f = X(BN x F) @ W(F x U)
// grid: (BN/128, U/16), block 256 (8 waves); each wave -> one 16x16 tile.
__global__ void gemm_f_kernel(const unsigned short* __restrict__ Xb,
                              const unsigned short* __restrict__ Wb,
                              unsigned short* __restrict__ Fb) {
    const int w    = threadIdx.x >> 5;
    const int m0   = blockIdx.x * 128 + w * 16;
    const int n0   = blockIdx.y * 16;
    const int lane = threadIdx.x & 31;
    const int half = lane >> 4;
    const int lm   = lane & 15;
    v8f acc = {};
#pragma unroll
    for (int k0 = 0; k0 < FF; k0 += 32) {
        const unsigned short* arow =
            Xb + (size_t)(m0 + lm) * FF + k0 + half * 8;
        v8bf lo = *reinterpret_cast<const v8bf*>(arow);
        v8bf hi = *reinterpret_cast<const v8bf*>(arow + 16);
        v16bf a = cat16(lo, hi);
        v16bf b;
#pragma unroll
        for (int s = 0; s < 16; ++s) {
            const int kb = k0 + 8 * half + s + (s >= 8 ? 8 : 0);
            b[s] = __builtin_bit_cast(__bf16, Wb[(size_t)kb * UU + n0 + lm]);
        }
        acc = __builtin_amdgcn_wmma_f32_16x16x32_bf16(false, a, false, b,
                                                      (short)0, acc, false, false);
    }
#pragma unroll
    for (int r = 0; r < 8; ++r) {
        const int m = m0 + r + 8 * half;
        Fb[(size_t)m * UU + n0 + lm] = f32_to_bf16(acc[r]);
    }
}

// ------------------- attn logits: attn[b,i,j] = (f·ai)[i,j] + (f·aj)[j,i] + mask
// grid: (N/64, N/32, B), block 256 = 8 waves arranged 4(i) x 2(j).
__global__ void attn_logits_kernel(const unsigned short* __restrict__ Fb,
                                   const unsigned short* __restrict__ Ab,  // ai bf16 [U,N]
                                   const unsigned short* __restrict__ Jb,  // aj bf16 [U,N]
                                   const float* __restrict__ adj,
                                   float* __restrict__ attn) {
    __shared__ float e2t[8][16][17];
    const int w    = threadIdx.x >> 5;
    const int wi   = w & 3;
    const int wj   = w >> 2;
    const int b    = blockIdx.z;
    const int gi0  = blockIdx.x * 64 + wi * 16;
    const int gj0  = blockIdx.y * 32 + wj * 16;
    const int lane = threadIdx.x & 31;
    const int half = lane >> 4;
    const int lm   = lane & 15;
    const unsigned short* fB = Fb + (size_t)b * NN * UU;

    v8f acc1 = {}, acc2 = {};
#pragma unroll
    for (int k0 = 0; k0 < UU; k0 += 32) {
        const int kb0 = k0 + 8 * half;
        {   // e1 tile: A rows = i-tile of f, B cols = j-tile of ai
            const unsigned short* arow = fB + (size_t)(gi0 + lm) * UU + k0 + half * 8;
            v16bf a = cat16(*reinterpret_cast<const v8bf*>(arow),
                            *reinterpret_cast<const v8bf*>(arow + 16));
            v16bf bb;
#pragma unroll
            for (int s = 0; s < 16; ++s) {
                const int kb = kb0 + s + (s >= 8 ? 8 : 0);
                bb[s] = __builtin_bit_cast(__bf16, Ab[(size_t)kb * NN + gj0 + lm]);
            }
            acc1 = __builtin_amdgcn_wmma_f32_16x16x32_bf16(false, a, false, bb,
                                                           (short)0, acc1, false, false);
        }
        {   // e2 tile (M=j, N=i): A rows = j-tile of f, B cols = i-tile of aj
            const unsigned short* arow = fB + (size_t)(gj0 + lm) * UU + k0 + half * 8;
            v16bf a = cat16(*reinterpret_cast<const v8bf*>(arow),
                            *reinterpret_cast<const v8bf*>(arow + 16));
            v16bf bb;
#pragma unroll
            for (int s = 0; s < 16; ++s) {
                const int kb = kb0 + s + (s >= 8 ? 8 : 0);
                bb[s] = __builtin_bit_cast(__bf16, Jb[(size_t)kb * NN + gi0 + lm]);
            }
            acc2 = __builtin_amdgcn_wmma_f32_16x16x32_bf16(false, a, false, bb,
                                                           (short)0, acc2, false, false);
        }
    }
    // transpose e2 through this wave's LDS region (same-wave DS ops are ordered)
#pragma unroll
    for (int r = 0; r < 8; ++r) e2t[w][r + 8 * half][lm] = acc2[r];
#pragma unroll
    for (int r = 0; r < 8; ++r) {
        const int il = r + 8 * half;   // local i of this output element
        const int jl = lm;             // local j
        const float e2 = e2t[w][jl][il];
        const int gi = gi0 + il, gj = gj0 + jl;
        const float maskv = adj[(size_t)gi * NN + gj] + (gi == gj ? 1.0f : 0.0f);
        attn[(size_t)b * NN * NN + (size_t)gi * NN + gj] =
            acc1[r] + e2 + NEGV * (1.0f - maskv);
    }
}

// --------------------- column softmax (axis=1): grid B*64 blocks, 256 threads
// block owns 64 columns; 4 row-chunks of 1024 reduced through LDS.
// Unroll-by-8 keeps ~8 independent loads in flight per lane (latency hiding).
__global__ void softmax_cols_kernel(float* __restrict__ attn) {
    __shared__ float red[4][64];
    const int tid = threadIdx.x;
    const int c   = tid & 63;
    const int rg  = tid >> 6;
    const int b   = blockIdx.x >> 6;
    const int j   = (blockIdx.x & 63) * 64 + c;
    float* A = attn + (size_t)b * NN * NN + j;
    const int i0 = rg * 1024, i1 = i0 + 1024;

    float m = -3.4e38f;
    for (int i = i0; i < i1; i += 8) {
        float x[8];
#pragma unroll
        for (int q = 0; q < 8; ++q) x[q] = A[(size_t)(i + q) * NN];
        float m01 = fmaxf(x[0], x[1]), m23 = fmaxf(x[2], x[3]);
        float m45 = fmaxf(x[4], x[5]), m67 = fmaxf(x[6], x[7]);
        m = fmaxf(m, fmaxf(fmaxf(m01, m23), fmaxf(m45, m67)));
    }
    red[rg][c] = m;
    __syncthreads();
    m = fmaxf(fmaxf(red[0][c], red[1][c]), fmaxf(red[2][c], red[3][c]));
    __syncthreads();

    float s = 0.0f;
    for (int i = i0; i < i1; i += 8) {
        float x[8];
#pragma unroll
        for (int q = 0; q < 8; ++q) x[q] = A[(size_t)(i + q) * NN];
#pragma unroll
        for (int q = 0; q < 8; ++q) s += __expf(x[q] - m);
    }
    red[rg][c] = s;
    __syncthreads();
    s = red[0][c] + red[1][c] + red[2][c] + red[3][c];
    const float inv = 1.0f / s;

    for (int i = i0; i < i1; i += 8) {
        float x[8];
#pragma unroll
        for (int q = 0; q < 8; ++q) x[q] = A[(size_t)(i + q) * NN];
#pragma unroll
        for (int q = 0; q < 8; ++q) A[(size_t)(i + q) * NN] = __expf(x[q] - m) * inv;
    }
}

// ---------------- out[b,j,u] = relu( sum_i attn[b,i,j] * f[b,i,u] )  (attn^T @ f)
// grid: (N/16, B), block 256 = 8 waves; wave w owns u-tile w.
// 64-deep i-chunks: one barrier pair per 2 WMMAs per wave.
__global__ void out_gemm_kernel(const float* __restrict__ attn,
                                const unsigned short* __restrict__ Fb,
                                float* __restrict__ out) {
    __shared__ __align__(16) unsigned short T[16 * 72];  // [j_local][i_local], pad 72
    const int b    = blockIdx.y;
    const int gj0  = blockIdx.x * 16;
    const int w    = threadIdx.x >> 5;
    const int u0   = w * 16;
    const int lane = threadIdx.x & 31;
    const int half = lane >> 4;
    const int lm   = lane & 15;
    const float* A = attn + (size_t)b * NN * NN;
    const unsigned short* fB = Fb + (size_t)b * NN * UU;

    v8f acc = {};
    for (int i0 = 0; i0 < NN; i0 += 64) {
        __syncthreads();
        {   // cooperative load 64(i) x 16(j) f32 -> LDS transposed bf16
            const int r  = threadIdx.x >> 2;        // 0..63 : i_local
            const int c0 = (threadIdx.x & 3) * 4;   // 0,4,8,12 : j_local quad
            const float* src = A + (size_t)(i0 + r) * NN + gj0 + c0;
            const float x0 = src[0], x1 = src[1], x2 = src[2], x3 = src[3];
            T[(c0 + 0) * 72 + r] = f32_to_bf16(x0);
            T[(c0 + 1) * 72 + r] = f32_to_bf16(x1);
            T[(c0 + 2) * 72 + r] = f32_to_bf16(x2);
            T[(c0 + 3) * 72 + r] = f32_to_bf16(x3);
        }
        __syncthreads();
#pragma unroll
        for (int kk = 0; kk < 64; kk += 32) {
            const unsigned short* arow = T + lm * 72 + kk + half * 8;
            v16bf a = cat16(*reinterpret_cast<const v8bf*>(arow),
                            *reinterpret_cast<const v8bf*>(arow + 16));
            v16bf bb;
#pragma unroll
            for (int s = 0; s < 16; ++s) {
                const int kb = kk + 8 * half + s + (s >= 8 ? 8 : 0);
                bb[s] = __builtin_bit_cast(__bf16, fB[(size_t)(i0 + kb) * UU + u0 + lm]);
            }
            acc = __builtin_amdgcn_wmma_f32_16x16x32_bf16(false, a, false, bb,
                                                          (short)0, acc, false, false);
        }
    }
#pragma unroll
    for (int r = 0; r < 8; ++r) {
        const int jl = r + 8 * half;
        out[(size_t)b * NN * UU + (size_t)(gj0 + jl) * UU + u0 + lm] =
            fmaxf(acc[r], 0.0f);
    }
}

// ----------------------------------------------------------------- launch
extern "C" void kernel_launch(void* const* d_in, const int* in_sizes, int n_in,
                              void* d_out, int out_size, void* d_ws, size_t ws_size,
                              hipStream_t stream) {
    const float* inputs = (const float*)d_in[0];  // [B,N,F]
    const float* w      = (const float*)d_in[1];  // [F,U]
    const float* ai     = (const float*)d_in[2];  // [U,N]
    const float* aj     = (const float*)d_in[3];  // [U,N]
    const float* adj    = (const float*)d_in[4];  // [N,N]

    float* outp  = (float*)d_out;                        // [B,N,U]
    float* attnp = outp + (size_t)BB * NN * UU;          // [B,N,N]

    char* ws = (char*)d_ws;
    unsigned short* Xb = (unsigned short*)(ws + 0);                 // 8 MiB
    unsigned short* Wb = (unsigned short*)(ws + 8388608);           // 64 KiB
    unsigned short* Ab = (unsigned short*)(ws + 8454144);           // 1 MiB
    unsigned short* Jb = (unsigned short*)(ws + 9502720);           // 1 MiB
    unsigned short* Fb = (unsigned short*)(ws + 10551296);          // 4 MiB

    // 1) f32 -> bf16 conversions
    cvt_bf16_kernel<<<2048, 256, 0, stream>>>(inputs, Xb, BB * NN * FF);
    cvt_bf16_kernel<<<128,  256, 0, stream>>>(w,      Wb, FF * UU);
    cvt_bf16_kernel<<<1024, 256, 0, stream>>>(ai,     Ab, UU * NN);
    cvt_bf16_kernel<<<1024, 256, 0, stream>>>(aj,     Jb, UU * NN);

    // 2) f = X @ W   (bf16 WMMA, f32 accum, stored bf16)
    gemm_f_kernel<<<dim3((BB * NN) / 128, UU / 16), 256, 0, stream>>>(Xb, Wb, Fb);

    // 3) attention logits (+mask) -> attn region of d_out
    attn_logits_kernel<<<dim3(NN / 64, NN / 32, BB), 256, 0, stream>>>(
        Fb, Ab, Jb, adj, attnp);

    // 4) softmax over axis=1 (columns), in place
    softmax_cols_kernel<<<BB * 64, 256, 0, stream>>>(attnp);

    // 5) out = relu(attn^T @ f)
    out_gemm_kernel<<<dim3(NN / 16, BB), 256, 0, stream>>>(attnp, Fb, outp);
}
